// PNALayer_87393994539136
// MI455X (gfx1250) — matrix operand order repaired and checked by reading, not compiled, not run on hardware
//
#include <hip/hip_runtime.h>
#include <hip/hip_bf16.h>
#include <math.h>

// ---------- problem constants ----------
#define NN 50000
#define EE 800000
#define FF 128
#define K1 256      // 2F   (edge GEMM K)
#define K2 1664     // 13F  (post GEMM K)
#define KT1 8       // K1/32
#define KT2 52      // K2/32
#define KT3 4       // FF/32
#define NT  8       // FF/16 output tiles
#define AVG_LOG 2.8332133440562162f
#define BN_EPS 1e-5f
#define STD_EPS 1e-5f

typedef __attribute__((ext_vector_type(16))) __bf16 v16bf;
typedef __attribute__((ext_vector_type(8)))  __bf16 v8bf;
typedef __attribute__((ext_vector_type(4)))  __bf16 v4bf;
typedef __attribute__((ext_vector_type(8)))  float  v8f;
typedef __attribute__((ext_vector_type(4)))  float  v4f;

static __device__ __forceinline__ v16bf cat16(v8bf lo, v8bf hi) {
  return __builtin_shufflevector(lo, hi, 0,1,2,3,4,5,6,7,8,9,10,11,12,13,14,15);
}

// float atomic min/max via order-preserving int mapping (init: +INF / -INF)
static __device__ __forceinline__ void atomicMinF(float* a, float v) {
  if (v >= 0.f) atomicMin((int*)a, __float_as_int(v));
  else          atomicMax((unsigned int*)a, __float_as_uint(v));
}
static __device__ __forceinline__ void atomicMaxF(float* a, float v) {
  if (v >= 0.f) atomicMax((int*)a, __float_as_int(v));
  else          atomicMin((unsigned int*)a, __float_as_uint(v));
}

// ---------- init: accumulators (vectorized b128 stores) ----------
__global__ void __launch_bounds__(256) k_init(float* __restrict__ sum, float* __restrict__ sumsq,
                                              float* __restrict__ mn, float* __restrict__ mx,
                                              float* __restrict__ cnt,
                                              float* __restrict__ bn_sum, float* __restrict__ bn_sq) {
  long long i = (long long)blockIdx.x * 256 + threadIdx.x;      // over NN*FF/4 vec4s
  if (i < (long long)NN * FF / 4) {
    v4f z = {0.f, 0.f, 0.f, 0.f};
    v4f pinf = {INFINITY, INFINITY, INFINITY, INFINITY};
    v4f ninf = {-INFINITY, -INFINITY, -INFINITY, -INFINITY};
    ((v4f*)sum)[i] = z; ((v4f*)sumsq)[i] = z;
    ((v4f*)mn)[i] = pinf; ((v4f*)mx)[i] = ninf;
  }
  if (i < NN) cnt[i] = 0.f;
  if (i < FF) { bn_sum[i] = 0.f; bn_sq[i] = 0.f; }
}

// ---------- h -> bf16 (4 elements/thread) ----------
__global__ void __launch_bounds__(256) k_hconv(const float* __restrict__ h, __bf16* __restrict__ hb) {
  long long i = (long long)blockIdx.x * 256 + threadIdx.x;      // over NN*FF/4
  if (i < (long long)NN * FF / 4) {
    v4f x = ((const v4f*)h)[i];
    v4bf y = { (__bf16)x.x, (__bf16)x.y, (__bf16)x.z, (__bf16)x.w };
    ((v4bf*)hb)[i] = y;
  }
}

// ---------- pack a KxFF weight (row-major) into per-lane WMMA B-fragments ----------
// frag layout: for (kt,nt): 32 lanes x 16 bf16 contiguous; lane holds column nt*16+(lane&15),
// K-locals: lanes 0-15 -> K 0..7,16..23, lanes 16-31 -> K 8..15,24..31 of the 32-wide k-tile.
__global__ void __launch_bounds__(256) k_pack(const float* __restrict__ W, __bf16* __restrict__ frag,
                                              int ktiles) {
  int idx = blockIdx.x * 256 + threadIdx.x;     // over ktiles*8*32
  if (idx >= ktiles * NT * 32) return;
  int lane = idx & 31;
  int nt   = (idx >> 5) & 7;
  int kt   = idx >> 8;
  int n    = nt * 16 + (lane & 15);
  int kb   = kt * 32 + ((lane >> 4) << 3);
  __bf16* o = frag + (long long)idx * 16;
  #pragma unroll
  for (int i = 0; i < 16; ++i) o[i] = (__bf16)W[(long long)(kb + (i & 7) + ((i >> 3) << 4)) * FF + n];
}

// ---------- per-dst degree ----------
__global__ void __launch_bounds__(256) k_count(const long long* __restrict__ dstI, float* __restrict__ cnt) {
  int e = blockIdx.x * 256 + threadIdx.x;
  if (e < EE) atomicAdd(&cnt[(int)dstI[e]], 1.f);
}

// ---------- edge GEMM (bf16 WMMA) + fused multi-aggregator scatter ----------
__global__ void __launch_bounds__(256) k_edge(const __bf16* __restrict__ hb,
                                              const long long* __restrict__ srcI,
                                              const long long* __restrict__ dstI,
                                              const __bf16* __restrict__ wpreF,
                                              const float* __restrict__ bpre,
                                              float* __restrict__ sum, float* __restrict__ sumsq,
                                              float* __restrict__ mn, float* __restrict__ mx) {
  extern __shared__ char smem[];
  __bf16* wlds = (__bf16*)smem;                  // 8*8*32*16 bf16 = 64KB
  // stage all W_pre fragments into LDS
  {
    const uint4* g = (const uint4*)wpreF;        // 65536B / 16 = 4096
    uint4* l = (uint4*)wlds;
    for (int i = threadIdx.x; i < 4096; i += 256) l[i] = g[i];
  }
  __syncthreads();

  const int wave = threadIdx.x >> 5;
  const int lane = threadIdx.x & 31;
  const int row  = lane & 15;
  const int hi   = lane >> 4;
  const long long tile = (long long)blockIdx.x * 8 + wave;   // exact: 50000 tiles
  const long long e0   = tile * 16;
  const int sL = (int)srcI[e0 + row];
  const int dL = (int)dstI[e0 + row];

  v8f acc[NT] = {};
  for (int kt = 0; kt < KT1; ++kt) {
    const int rbase = (kt < 4) ? dL : sL;                    // first half x_i=h[dst], then x_j=h[src]
    const int koff  = (kt & 3) * 32 + hi * 8;
    const __bf16* ap = hb + (long long)rbase * FF + koff;
    v16bf a = cat16(*(const v8bf*)ap, *(const v8bf*)(ap + 16));
    #pragma unroll
    for (int n = 0; n < NT; ++n) {
      const __bf16* bp = wlds + ((kt * NT + n) * 32 + lane) * 16;
      v16bf b = cat16(*(const v8bf*)bp, *(const v8bf*)(bp + 8));
      acc[n] = __builtin_amdgcn_wmma_f32_16x16x32_bf16(false, a, false, b,
                                                       (short)0, acc[n], false, false);
    }
  }

  // bias + scatter (sum / sumsq / min / max)
  #pragma unroll
  for (int n = 0; n < NT; ++n) {
    const int col = n * 16 + row;
    const float bias = bpre[col];
    #pragma unroll
    for (int r = 0; r < 8; ++r) {
      const int m = hi * 8 + r;                  // output row of this acc element
      const int d = __shfl(dL, m, 32);           // dst of edge e0+m
      const float v = acc[n][r] + bias;
      const long long off = (long long)d * FF + col;
      atomicAdd(sum + off, v);
      atomicAdd(sumsq + off, v * v);
      atomicMinF(mn + off, v);
      atomicMaxF(mx + off, v);
    }
  }
}

// ---------- node update: build 13F row, WMMA vs W_post, WMMA vs W_lin, BN stats ----------
__global__ void __launch_bounds__(32) k_node(const float* __restrict__ h,
                                             const float* __restrict__ sum, const float* __restrict__ sumsq,
                                             const float* __restrict__ mn, const float* __restrict__ mx,
                                             const float* __restrict__ cnt,
                                             const __bf16* __restrict__ wpostF, const float* __restrict__ bpost,
                                             const __bf16* __restrict__ wlinF,  const float* __restrict__ blin,
                                             float* __restrict__ preBN,
                                             float* __restrict__ bn_sum, float* __restrict__ bn_sq) {
  extern __shared__ char smem[];
  __bf16* x13 = (__bf16*)smem;                  // [16][1664] = 53248B
  __bf16* tt  = x13 + 16 * K2;                  // [16][128]  = 4096B
  const int lane = threadIdx.x;
  const int row = lane & 15, hi = lane >> 4;
  const long long n0 = (long long)blockIdx.x * 16;   // exact: 3125 tiles

  // build [h | aggr | aggr*amp | aggr*att] rows in LDS (bf16).
  // Iterate the (row, col) space once: compute the 4 aggregates + 2 degree
  // scalers per element, fan out all 13 writes. All 32 lanes of an iteration
  // hit the same node -> cnt/deg wave-uniform and cache-hot.
  for (int idx = lane; idx < 16 * FF; idx += 32) {
    const int rr = idx >> 7;            // node row in tile
    const int c  = idx & 127;           // feature column
    const long long node = n0 + rr;
    const long long off  = node * FF + c;

    const float cv  = cnt[node];
    const float deg = fmaxf(cv, 1.f);
    const float inv = 1.f / deg;
    const float me  = sum[off] * inv;
    const float ms  = sumsq[off] * inv;
    const float sd  = sqrtf(fmaxf(ms - me * me, 0.f) + STD_EPS);
    const float lo  = (cv > 0.f) ? mn[off] : 0.f;
    const float hi_ = (cv > 0.f) ? mx[off] : 0.f;
    const float ld  = logf(deg + 1.f);
    const float amp = ld * (1.f / AVG_LOG);
    const float att = AVG_LOG / ld;

    __bf16* rp = x13 + rr * K2;
    rp[c] = (__bf16)h[off];                       // self features
    #pragma unroll
    for (int seg = 0; seg < 3; ++seg) {
      const float s = (seg == 0) ? 1.f : ((seg == 1) ? amp : att);
      __bf16* sp = rp + FF + seg * (4 * FF);
      sp[0 * FF + c] = (__bf16)(me  * s);
      sp[1 * FF + c] = (__bf16)(lo  * s);
      sp[2 * FF + c] = (__bf16)(hi_ * s);
      sp[3 * FF + c] = (__bf16)(sd  * s);
    }
  }
  __syncthreads();   // single-wave block: cheap (S_NOP barrier), guarantees LDS visibility

  // GEMM1: [16 x 1664] @ W_post -> [16 x 128]
  v8f acc[NT] = {};
  for (int kt = 0; kt < KT2; ++kt) {
    const __bf16* ap = x13 + row * K2 + kt * 32 + hi * 8;
    v16bf a = cat16(*(const v8bf*)ap, *(const v8bf*)(ap + 16));
    #pragma unroll
    for (int n = 0; n < NT; ++n) {
      const __bf16* bp = wpostF + ((long long)(kt * NT + n) * 32 + lane) * 16;
      v16bf b = cat16(*(const v8bf*)bp, *(const v8bf*)(bp + 8));
      acc[n] = __builtin_amdgcn_wmma_f32_16x16x32_bf16(false, a, false, b,
                                                       (short)0, acc[n], false, false);
    }
  }

  // bias + D-layout -> A-layout transpose through LDS (bf16)
  #pragma unroll
  for (int n = 0; n < NT; ++n) {
    const int col = n * 16 + row;
    const float bias = bpost[col];
    #pragma unroll
    for (int r = 0; r < 8; ++r) {
      const int m = hi * 8 + r;
      tt[m * FF + col] = (__bf16)(acc[n][r] + bias);
    }
  }
  __syncthreads();

  // GEMM2: [16 x 128] @ W_lin -> [16 x 128]
  v8f a2[NT] = {};
  for (int kt = 0; kt < KT3; ++kt) {
    const __bf16* ap = tt + row * FF + kt * 32 + hi * 8;
    v16bf a = cat16(*(const v8bf*)ap, *(const v8bf*)(ap + 16));
    #pragma unroll
    for (int n = 0; n < NT; ++n) {
      const __bf16* bp = wlinF + ((long long)(kt * NT + n) * 32 + lane) * 16;
      v16bf b = cat16(*(const v8bf*)bp, *(const v8bf*)(bp + 8));
      a2[n] = __builtin_amdgcn_wmma_f32_16x16x32_bf16(false, a, false, b,
                                                      (short)0, a2[n], false, false);
    }
  }

  // bias, store pre-BN, wave-reduced BN column statistics
  #pragma unroll
  for (int n = 0; n < NT; ++n) {
    const int col = n * 16 + row;
    const float bias = blin[col];
    float s = 0.f, s2 = 0.f;
    #pragma unroll
    for (int r = 0; r < 8; ++r) {
      const int m = hi * 8 + r;
      const long long node = n0 + m;
      const float v = a2[n][r] + bias;
      preBN[node * FF + col] = v;
      s += v; s2 += v * v;
    }
    atomicAdd(bn_sum + col, s);
    atomicAdd(bn_sq  + col, s2);
  }
}

// ---------- BN + ReLU + residual ----------
__global__ void __launch_bounds__(256) k_final(const float* __restrict__ preBN,
                                               const float* __restrict__ bn_sum, const float* __restrict__ bn_sq,
                                               const float* __restrict__ gamma, const float* __restrict__ beta,
                                               const float* __restrict__ h, float* __restrict__ out) {
  long long i = (long long)blockIdx.x * 256 + threadIdx.x;
  if (i >= (long long)NN * FF) return;
  int f = (int)(i & (FF - 1));
  float mu  = bn_sum[f] * (1.f / NN);
  float var = bn_sq[f] * (1.f / NN) - mu * mu;
  float x = (preBN[i] - mu) * rsqrtf(var + BN_EPS) * gamma[f] + beta[f];
  out[i] = fmaxf(x, 0.f) + h[i];
}

// ---------- host launcher ----------
extern "C" void kernel_launch(void* const* d_in, const int* in_sizes, int n_in,
                              void* d_out, int out_size, void* d_ws, size_t ws_size,
                              hipStream_t stream) {
  const float*     h     = (const float*)d_in[0];
  const long long* ei    = (const long long*)d_in[1];
  const float*     Wpre  = (const float*)d_in[2];
  const float*     bpre  = (const float*)d_in[3];
  const float*     Wpost = (const float*)d_in[4];
  const float*     bpost = (const float*)d_in[5];
  const float*     Wlin  = (const float*)d_in[6];
  const float*     blin  = (const float*)d_in[7];
  const float*     gam   = (const float*)d_in[8];
  const float*     bet   = (const float*)d_in[9];
  const long long* srcI  = ei;        // edge_index[0]
  const long long* dstI  = ei + EE;   // edge_index[1]

  char* ws = (char*)d_ws;
  size_t off = 0;
  auto alloc = [&](size_t bytes) -> char* {
    char* p = ws + off;
    off += (bytes + 255) & ~(size_t)255;
    return p;
  };
  __bf16* hb     = (__bf16*)alloc((size_t)NN * FF * 2);
  __bf16* wpreF  = (__bf16*)alloc((size_t)KT1 * NT * 32 * 16 * 2);
  __bf16* wpostF = (__bf16*)alloc((size_t)KT2 * NT * 32 * 16 * 2);
  __bf16* wlinF  = (__bf16*)alloc((size_t)KT3 * NT * 32 * 16 * 2);
  float*  sum    = (float*)alloc((size_t)NN * FF * 4);
  float*  sumsq  = (float*)alloc((size_t)NN * FF * 4);
  float*  mnb    = (float*)alloc((size_t)NN * FF * 4);
  float*  mxb    = (float*)alloc((size_t)NN * FF * 4);
  float*  cnt    = (float*)alloc((size_t)NN * 4);
  float*  preBN  = (float*)alloc((size_t)NN * FF * 4);
  float*  bn_sum = (float*)alloc((size_t)FF * 4);
  float*  bn_sq  = (float*)alloc((size_t)FF * 4);
  (void)ws_size; (void)in_sizes; (void)n_in; (void)out_size;

  const int v4_blocks = (NN * FF / 4 + 255) / 256;   // 6250
  const int nf_blocks = (NN * FF + 255) / 256;       // 25000

  k_init <<<v4_blocks, 256, 0, stream>>>(sum, sumsq, mnb, mxb, cnt, bn_sum, bn_sq);
  k_hconv<<<v4_blocks, 256, 0, stream>>>(h, hb);
  k_pack <<<(KT1 * NT * 32 + 255) / 256, 256, 0, stream>>>(Wpre,  wpreF,  KT1);
  k_pack <<<(KT2 * NT * 32 + 255) / 256, 256, 0, stream>>>(Wpost, wpostF, KT2);
  k_pack <<<(KT3 * NT * 32 + 255) / 256, 256, 0, stream>>>(Wlin,  wlinF,  KT3);
  k_count<<<(EE + 255) / 256, 256, 0, stream>>>(dstI, cnt);

  // 50000 edge tiles, 8 waves/block -> 6250 blocks; 64KB LDS for W_pre fragments
  k_edge<<<EE / 16 / 8, 256, 65536, stream>>>(hb, srcI, dstI, wpreF, bpre,
                                              sum, sumsq, mnb, mxb);

  // 3125 node tiles, 1 wave/block; LDS = 16*1664*2 + 16*128*2 = 57344B
  k_node<<<NN / 16, 32, 16 * K2 * 2 + 16 * FF * 2, stream>>>(
      h, sum, sumsq, mnb, mxb, cnt, wpostF, bpost, wlinF, blin,
      preBN, bn_sum, bn_sq);

  k_final<<<nf_blocks, 256, 0, stream>>>(preBN, bn_sum, bn_sq, gam, bet, h, (float*)d_out);
}
